// SPELayer_80985903333703
// MI455X (gfx1250) — compile-verified, matrix operand order checked by reading, not run on hardware
//
#include <hip/hip_runtime.h>
#include <hip/hip_bf16.h>

// ---------------------------------------------------------------------------
// Types for CDNA5 WMMA (wave32)
// ---------------------------------------------------------------------------
typedef __attribute__((ext_vector_type(16))) __bf16 v16bf;
typedef __attribute__((ext_vector_type(8)))  float  v8f;
typedef unsigned short u16;
typedef unsigned int   u32;
typedef unsigned long long u64;

union FragAB { v16bf v; uint4 q[2]; };

__device__ __forceinline__ u16 f2bf(float f) {
    u32 a = __float_as_uint(f);
    return (u16)((a + 0x7FFFu + ((a >> 16) & 1u)) >> 16);   // RNE f32 -> bf16
}

// ---------------------------------------------------------------------------
// Tiled GEMM:  C[M,N] = act( A[M,K] * W[N,K]^T + bias ) + residual
// A, W are bf16 in memory. Double-buffered LDS pipeline: tile k+1 staged with
// GLOBAL_LOAD_ASYNC_TO_LDS_B128 (ASYNCcnt) while tile k is consumed by
// v_wmma_f32_16x16x32_bf16 (f32 accumulation). One barrier per K-step.
// Block tile 128x128, BK=32, 8 waves (4 in M x 2 in N), wave tile 32x64.
// ---------------------------------------------------------------------------
#define BM 128
#define BN 128
#define BK 32
#define SROW 48   // LDS row stride in halfs (96B: 16B-aligned, 24-bank stride)

__device__ __forceinline__ void async_copy32B(u32 lds_addr, u64 gaddr) {
    // copies 32B: offset applies to BOTH lds and global address (ISA 8.2)
    asm volatile("global_load_async_to_lds_b128 %0, %1, off"
                 :: "v"(lds_addr), "v"(gaddr) : "memory");
    asm volatile("global_load_async_to_lds_b128 %0, %1, off offset:16"
                 :: "v"(lds_addr), "v"(gaddr) : "memory");
}

__global__ __launch_bounds__(256)
void gemm_bf16_wmma(const u16* __restrict__ A, int lda,
                    const u16* __restrict__ W, int ldw,
                    float* __restrict__ C, int ldc,
                    u16* __restrict__ Cbf, int ldcb,
                    const float* __restrict__ bias,
                    const float* __restrict__ residual, int ldr,
                    int M, int N, int K, int act)
{
    __shared__ __align__(16) u16 sA[2][BM * SROW];
    __shared__ __align__(16) u16 sW[2][BN * SROW];

    const int tid  = threadIdx.x;
    const int lane = tid & 31;
    const int wave = tid >> 5;
    const int wm   = wave & 3;   // wave M offset (x32)
    const int wn   = wave >> 2;  // wave N offset (x64)
    const int blockM = blockIdx.y * BM;
    const int blockN = blockIdx.x * BN;

    v8f acc[2][4];
    {
        v8f zero = {};
        #pragma unroll
        for (int i = 0; i < 2; ++i)
            #pragma unroll
            for (int j = 0; j < 4; ++j) acc[i][j] = zero;
    }

    // staging map: 2 threads per row, each thread copies 16 halfs (32B)
    const int ldRow  = tid >> 1;          // 0..127
    const int ldColH = (tid & 1) * 16;    // 0 or 16 (halfs)

    const int  gr = blockM + ldRow;
    const int  gn = blockN + ldRow;
    const bool aOk = (gr < M);
    const bool wOk = (gn < N);
    const u16* srcA = A + (size_t)gr * lda + ldColH;
    const u16* srcW = W + (size_t)gn * ldw + ldColH;
    const int  stOff = ldRow * SROW + ldColH;
    const u32  dstA0 = (u32)(size_t)&sA[0][stOff];
    const u32  dstA1 = (u32)(size_t)&sA[1][stOff];
    const u32  dstW0 = (u32)(size_t)&sW[0][stOff];
    const u32  dstW1 = (u32)(size_t)&sW[1][stOff];

    // prologue: OOB rows stay zero forever (DMA never touches them)
    if (!aOk) {
        uint4 z = {0u, 0u, 0u, 0u};
        *(uint4*)&sA[0][stOff] = z; *(uint4*)&sA[0][stOff + 8] = z;
        *(uint4*)&sA[1][stOff] = z; *(uint4*)&sA[1][stOff + 8] = z;
    }
    if (!wOk) {
        uint4 z = {0u, 0u, 0u, 0u};
        *(uint4*)&sW[0][stOff] = z; *(uint4*)&sW[0][stOff + 8] = z;
        *(uint4*)&sW[1][stOff] = z; *(uint4*)&sW[1][stOff + 8] = z;
    }
    // issue tile 0 into buffer 0
    if (aOk) async_copy32B(dstA0, (u64)(size_t)srcA);
    if (wOk) async_copy32B(dstW0, (u64)(size_t)srcW);

    int buf = 0;
    for (int k0 = 0; k0 < K; k0 += BK) {
        asm volatile("s_wait_asynccnt 0x0" ::: "memory");  // tile k0 copies done
        __syncthreads();   // all waves: tile k0 staged, tile k0-BK fully consumed

        if (k0 + BK < K) {  // stage tile k0+BK into the other buffer (overlaps WMMA)
            if (aOk) async_copy32B(buf ? dstA0 : dstA1,
                                   (u64)(size_t)(srcA + k0 + BK));
            if (wOk) async_copy32B(buf ? dstW0 : dstW1,
                                   (u64)(size_t)(srcW + k0 + BK));
            if (k0 + 2 * BK < K) {       // warm L2 for the tile after next
                __builtin_prefetch(srcA + k0 + 2 * BK, 0, 1);
                __builtin_prefetch(srcW + k0 + 2 * BK, 0, 1);
            }
        }

        // Fragment K chunks per ISA layout:
        //   lane<16 : K {0..7, 16..23};  lane>=16 : K {8..15, 24..31}
        const int c0 = (lane < 16) ? 0 : 8;
        const int ml = lane & 15;
        const u16* bA = sA[buf];
        const u16* bW = sW[buf];

        FragAB aFrag[2];
        #pragma unroll
        for (int i = 0; i < 2; ++i) {
            int r = wm * 32 + i * 16 + ml;
            aFrag[i].q[0] = *(const uint4*)&bA[r * SROW + c0];
            aFrag[i].q[1] = *(const uint4*)&bA[r * SROW + c0 + 16];
        }
        #pragma unroll
        for (int j = 0; j < 4; ++j) {
            int n = wn * 64 + j * 16 + ml;
            FragAB bFrag;
            bFrag.q[0] = *(const uint4*)&bW[n * SROW + c0];
            bFrag.q[1] = *(const uint4*)&bW[n * SROW + c0 + 16];
            #pragma unroll
            for (int i = 0; i < 2; ++i) {
                acc[i][j] = __builtin_amdgcn_wmma_f32_16x16x32_bf16(
                    false, aFrag[i].v, false, bFrag.v,
                    (short)0, acc[i][j], false, false);
            }
        }
        buf ^= 1;
    }

    // Epilogue. C/D layout: VGPR r -> M = r (lanes 0..15) / M = 8+r (lanes 16..31)
    const int halfSel = (lane < 16) ? 0 : 8;
    const int colLane = lane & 15;
    #pragma unroll
    for (int i = 0; i < 2; ++i) {
        #pragma unroll
        for (int j = 0; j < 4; ++j) {
            int col = blockN + wn * 64 + j * 16 + colLane;
            if (col >= N) continue;
            float bv = bias ? bias[col] : 0.f;
            #pragma unroll
            for (int r = 0; r < 8; ++r) {
                int row = blockM + wm * 32 + i * 16 + r + halfSel;
                if (row >= M) continue;
                float v = acc[i][j][r] + bv;
                if (act == 1) v = (v > 20.f) ? v : __logf(1.f + __expf(v)); // softplus
                if (residual) v += residual[(size_t)row * ldr + col];
                C[(size_t)row * ldc + col] = v;
                if (Cbf) Cbf[(size_t)row * ldcb + col] = f2bf(v);
            }
        }
    }
}

// ---------------------------------------------------------------------------
// f32 -> bf16 bulk conversion (weights, once per launch)
// ---------------------------------------------------------------------------
__global__ __launch_bounds__(256)
void cvt_bf16_k(const float* __restrict__ src, u16* __restrict__ dst, int n)
{
    int i = blockIdx.x * 256 + threadIdx.x;
    if (i < n) dst[i] = f2bf(src[i]);
}

// ---------------------------------------------------------------------------
// LayerNorm: one block per row; bf16 output (consumer is a GEMM)
// ---------------------------------------------------------------------------
__global__ __launch_bounds__(256)
void layernorm_k(const float* __restrict__ x, const float* __restrict__ g,
                 const float* __restrict__ b, u16* __restrict__ out, int cols)
{
    const int row = blockIdx.x;
    const int tid = threadIdx.x;
    const float* xr = x + (size_t)row * cols;
    float s = 0.f, s2 = 0.f;
    for (int c = tid; c < cols; c += 256) { float v = xr[c]; s += v; s2 += v * v; }
    __shared__ float sh1[256], sh2[256];
    sh1[tid] = s; sh2[tid] = s2;
    __syncthreads();
    for (int st = 128; st > 0; st >>= 1) {
        if (tid < st) { sh1[tid] += sh1[tid + st]; sh2[tid] += sh2[tid + st]; }
        __syncthreads();
    }
    float inv = 1.f / (float)cols;
    float mu  = sh1[0] * inv;
    float var = sh2[0] * inv - mu * mu;
    float rs  = rsqrtf(var + 1e-5f);
    u16* orow = out + (size_t)row * cols;
    for (int c = tid; c < cols; c += 256)
        orow[c] = f2bf((xr[c] - mu) * rs * g[c] + b[c]);
}

// ---------------------------------------------------------------------------
// Depthwise causal conv (width 4) + bias + SiLU.  Writes f32 (scan) + bf16 (GEMM)
// ---------------------------------------------------------------------------
__global__ __launch_bounds__(256)
void conv_silu_k(const float* __restrict__ xz, const float* __restrict__ cw,
                 const float* __restrict__ cb, float* __restrict__ xc,
                 u16* __restrict__ xcb, int L, int Di, int ldxz, int total)
{
    int idx = blockIdx.x * 256 + threadIdx.x;
    if (idx >= total) return;
    int d = idx & (Di - 1);
    int l = (idx / Di) % L;
    int b = idx / (Di * L);
    const float* xp = xz + (size_t)b * L * ldxz + d;
    float acc = cb[d];
    #pragma unroll
    for (int k = 0; k < 4; ++k) {
        int ls = l - 3 + k;
        if (ls >= 0) acc = fmaf(cw[d * 4 + k], xp[(size_t)ls * ldxz], acc);
    }
    float sig = 1.f / (1.f + __expf(-acc));
    float v = acc * sig;
    size_t o = (size_t)(b * L + l) * Di + d;
    xc[o]  = v;
    xcb[o] = f2bf(v);
}

// ---------------------------------------------------------------------------
// Selective scan. One lane per (b, d, n); 16 state lanes reduced via shuffles.
// Next-step operands prefetched into registers so loads overlap the exp chain.
// Fuses: dA/dBu recompute, recurrence, y = h.C, skip (+xc*D), SiLU(z) gate.
// ---------------------------------------------------------------------------
__global__ __launch_bounds__(256)
void scan_k(const float* __restrict__ delta, const float* __restrict__ xdbl,
            const float* __restrict__ xc,    const float* __restrict__ xz,
            const float* __restrict__ A_log, const float* __restrict__ Dp,
            u16* __restrict__ yg, int L, int Di, int ldx, int ldxz)
{
    int g  = blockIdx.x * 256 + threadIdx.x;
    int n  = g & 15;
    int ch = g >> 4;
    int d  = ch & (Di - 1);
    int b  = ch / Di;
    float a  = -__expf(A_log[d * 16 + n]);
    float Dv = Dp[d];
    float h  = 0.f;

    size_t row0 = (size_t)b * L;
    float dl  = delta[row0 * Di + d];
    float xcv = xc[row0 * Di + d];
    float Bv  = xdbl[row0 * ldx + 64 + n];
    float Cv  = xdbl[row0 * ldx + 80 + n];
    float zv  = xz[row0 * ldxz + Di + d];

    for (int l = 0; l < L; ++l) {
        // prefetch next step (independent of the recurrence chain)
        float dl_n = 0.f, xcv_n = 0.f, Bv_n = 0.f, Cv_n = 0.f, zv_n = 0.f;
        if (l + 1 < L) {
            size_t t2 = row0 + l + 1;
            dl_n  = delta[t2 * Di + d];
            xcv_n = xc[t2 * Di + d];
            Bv_n  = xdbl[t2 * ldx + 64 + n];
            Cv_n  = xdbl[t2 * ldx + 80 + n];
            zv_n  = xz[t2 * ldxz + Di + d];
        }
        h = __expf(dl * a) * h + dl * Bv * xcv;
        float contrib = h * Cv;
        contrib += __shfl_xor(contrib, 1, 32);
        contrib += __shfl_xor(contrib, 2, 32);
        contrib += __shfl_xor(contrib, 4, 32);
        contrib += __shfl_xor(contrib, 8, 32);
        if (n == 0) {
            float y  = contrib + xcv * Dv;
            float sz = zv / (1.f + __expf(-zv));
            yg[(row0 + l) * Di + d] = f2bf(y * sz);
        }
        dl = dl_n; xcv = xcv_n; Bv = Bv_n; Cv = Cv_n; zv = zv_n;
    }
}

// ---------------------------------------------------------------------------
// FFN gate: gated = silu(gv[:, :dm]) * gv[:, dm:]  -> bf16 (consumer is a GEMM)
// ---------------------------------------------------------------------------
__global__ __launch_bounds__(256)
void gate_k(const float* __restrict__ gv, u16* __restrict__ gated,
            int dm, int total)
{
    int idx = blockIdx.x * 256 + threadIdx.x;
    if (idx >= total) return;
    int m = idx / dm;
    int j = idx - m * dm;
    float gt = gv[(size_t)m * 2 * dm + j];
    float vl = gv[(size_t)m * 2 * dm + dm + j];
    float s  = gt / (1.f + __expf(-gt));
    gated[idx] = f2bf(s * vl);
}

// ---------------------------------------------------------------------------
// Launcher
// ---------------------------------------------------------------------------
extern "C" void kernel_launch(void* const* d_in, const int* in_sizes, int n_in,
                              void* d_out, int out_size, void* d_ws, size_t ws_size,
                              hipStream_t stream)
{
    const float* x        = (const float*)d_in[0];
    const float* n1_g     = (const float*)d_in[1];
    const float* n1_b     = (const float*)d_in[2];
    const float* n2_g     = (const float*)d_in[3];
    const float* n2_b     = (const float*)d_in[4];
    const float* in_projW = (const float*)d_in[5];
    const float* conv_w   = (const float*)d_in[6];
    const float* conv_b   = (const float*)d_in[7];
    const float* x_projW  = (const float*)d_in[8];
    const float* dt_w     = (const float*)d_in[9];
    const float* dt_b     = (const float*)d_in[10];
    const float* A_log    = (const float*)d_in[11];
    const float* Dp       = (const float*)d_in[12];
    const float* out_w    = (const float*)d_in[13];
    const float* ffg_w    = (const float*)d_in[14];
    const float* ffg_b    = (const float*)d_in[15];
    const float* ffp_w    = (const float*)d_in[16];
    const float* ffp_b    = (const float*)d_in[17];
    float* outp = (float*)d_out;

    const int Bsz = 2, L = 1024, DM = 1024, DI = 2048;
    const int Mr  = Bsz * L;     // 2048 rows
    const int LDX = 96;          // x_dbl row stride (dt 0..63, B 64..79, C 80..95)

    // ---------------- workspace layout ----------------
    float* wsf   = (float*)d_ws;
    float* xz    = wsf;                               // 2048*4096 f32
    float* xc    = xz    + (size_t)Mr * 2 * DI;       // 2048*2048 f32
    float* xdbl  = xc    + (size_t)Mr * DI;           // 2048*96   f32
    float* delta = xdbl  + (size_t)Mr * LDX;          // 2048*2048 f32
    float* x1    = delta + (size_t)Mr * DI;           // 2048*1024 f32
    float* gv    = xz;   // reuse after scan consumed z (needs 2048*2048)

    u16* wsb   = (u16*)(x1 + (size_t)Mr * DM);
    u16* nb    = wsb;                                 // 2048*1024 bf16 (ln1/ln2)
    u16* xcb   = nb    + (size_t)Mr * DM;             // 2048*2048 bf16
    u16* xdblb = xcb   + (size_t)Mr * DI;             // 2048*96   bf16
    u16* ygb   = xdblb + (size_t)Mr * LDX;            // 2048*2048 bf16
    u16* wip   = ygb   + (size_t)Mr * DI;             // 4096*1024 bf16 weights
    u16* wxp   = wip   + (size_t)(2 * DI) * DM;       // 96*2048
    u16* wdt   = wxp   + (size_t)LDX * DI;            // 2048*64
    u16* wout  = wdt   + (size_t)DI * 64;             // 1024*2048
    u16* wffg  = wout  + (size_t)DM * DI;             // 2048*1024
    u16* wffp  = wffg  + (size_t)(2 * DM) * DM;       // 1024*1024
    u16* gatedb = xcb;  // reuse after x_proj GEMM    // 2048*1024 bf16

    auto cvt = [&](const float* s, u16* dst, int n) {
        cvt_bf16_k<<<(n + 255) / 256, 256, 0, stream>>>(s, dst, n);
    };
    auto gemm = [&](const u16* A, int lda, const u16* W, int ldw,
                    float* C, int ldc, u16* Cbf, int ldcb,
                    const float* bias, const float* res, int ldr,
                    int M, int N, int K, int act) {
        dim3 grid((N + BN - 1) / BN, (M + BM - 1) / BM);
        gemm_bf16_wmma<<<grid, 256, 0, stream>>>(A, lda, W, ldw, C, ldc,
                                                 Cbf, ldcb, bias, res, ldr,
                                                 M, N, K, act);
    };

    // 0. weights -> bf16 (once per launch; ~20MB, negligible at 23.3 TB/s)
    cvt(in_projW, wip,  2 * DI * DM);
    cvt(x_projW,  wxp,  LDX * DI);
    cvt(dt_w,     wdt,  DI * 64);
    cvt(out_w,    wout, DM * DI);
    cvt(ffg_w,    wffg, 2 * DM * DM);
    cvt(ffp_w,    wffp, DM * DM);

    // 1. nb = LN(x; n1)  (bf16)
    layernorm_k<<<Mr, 256, 0, stream>>>(x, n1_g, n1_b, nb, DM);

    // 2. xz = nb @ in_proj^T                [2048 x 4096] f32
    gemm(nb, DM, wip, DM, xz, 2 * DI, nullptr, 0, nullptr, nullptr, 0,
         Mr, 2 * DI, DM, 0);

    // 3. xc/xcb = silu(causal_dwconv(x_in) + conv_b)
    {
        int total = Bsz * L * DI;
        conv_silu_k<<<(total + 255) / 256, 256, 0, stream>>>(
            xz, conv_w, conv_b, xc, xcb, L, DI, 2 * DI, total);
    }

    // 4. x_dbl = xcb @ x_proj^T             [2048 x 96] f32 + bf16 copy
    gemm(xcb, DI, wxp, DI, xdbl, LDX, xdblb, LDX, nullptr, nullptr, 0,
         Mr, LDX, DI, 0);

    // 5. delta = softplus(x_dbl[:, :64] @ dt_w^T + dt_b)   [2048 x 2048] f32
    gemm(xdblb, LDX, wdt, 64, delta, DI, nullptr, 0, dt_b, nullptr, 0,
         Mr, DI, 64, 1);

    // 6. selective scan + skip + gate -> ygb (bf16)
    {
        int total = Bsz * DI * 16;   // 65536 lanes
        scan_k<<<total / 256, 256, 0, stream>>>(
            delta, xdbl, xc, xz, A_log, Dp, ygb, L, DI, LDX, 2 * DI);
    }

    // 7. x1 = x + ygb @ out_w^T             [2048 x 1024] f32
    gemm(ygb, DI, wout, DI, x1, DM, nullptr, 0, nullptr, x, DM,
         Mr, DM, DI, 0);

    // 8. nb = LN(x1; n2)  (buffer reuse)
    layernorm_k<<<Mr, 256, 0, stream>>>(x1, n2_g, n2_b, nb, DM);

    // 9. gv = nb @ ffg_w^T + ffg_b          [2048 x 2048] f32 (reuses xz)
    gemm(nb, DM, wffg, DM, gv, 2 * DM, nullptr, 0, ffg_b, nullptr, 0,
         Mr, 2 * DM, DM, 0);

    // 10. gatedb = silu(gate) * val         [2048 x 1024] bf16 (reuses xcb)
    {
        int total = Mr * DM;
        gate_k<<<(total + 255) / 256, 256, 0, stream>>>(gv, gatedb, DM, total);
    }

    // 11. out = x1 + gatedb @ ffp_w^T + ffp_b
    gemm(gatedb, DM, wffp, DM, outp, DM, nullptr, 0, ffp_b, x1, DM,
         Mr, DM, DM, 0);

    (void)in_sizes; (void)n_in; (void)out_size; (void)ws_size;
}
